// LSTMRegressor_10222022165236
// MI455X (gfx1250) — matrix operand order repaired and checked by reading, not compiled, not run on hardware
//
#include <hip/hip_runtime.h>
#include <hip/hip_bf16.h>

typedef __attribute__((ext_vector_type(16))) _Float16 v16h;
typedef __attribute__((ext_vector_type(8)))  float    v8f;
typedef __attribute__((ext_vector_type(4)))  int      v4i;

union FragH { v16h v; float4 f4[2]; };
union AccF  { v8f  v; float  f[8]; };

#define WMMA_F16(a, b, c) \
  __builtin_amdgcn_wmma_f32_16x16x32_f16(false, (a), false, (b), (short)0, (c), false, false)

// ---------------- async global->LDS copy (CDNA5 ASYNCcnt path) --------------
#if defined(__has_builtin)
# if __has_builtin(__builtin_amdgcn_global_load_async_to_lds_b128)
#  define HAVE_ASYNC_LDS 1
# endif
#endif
#ifndef HAVE_ASYNC_LDS
# define HAVE_ASYNC_LDS 0
#endif

__device__ __forceinline__ void async_copy16(const _Float16* g, _Float16* l) {
#if HAVE_ASYNC_LDS
  typedef __attribute__((address_space(1))) v4i* gp_t;   // global 128-bit
  typedef __attribute__((address_space(3))) v4i* lp_t;   // LDS 128-bit
  __builtin_amdgcn_global_load_async_to_lds_b128((gp_t)g, (lp_t)l, 0, 0);
#else
  *(float4*)l = *(const float4*)g;  // synchronous fallback
#endif
}

__device__ __forceinline__ void async_wait0() {
#if HAVE_ASYNC_LDS
# if defined(__has_builtin) && __has_builtin(__builtin_amdgcn_s_wait_asynccnt)
  __builtin_amdgcn_s_wait_asynccnt(0);
# else
  asm volatile("s_wait_asynccnt 0x0" ::: "memory");
# endif
#endif
}

// ---------------------------------------------------------------- converts
__global__ __launch_bounds__(256) void cvt_f32_to_f16_k(const float* __restrict__ in,
                                                        _Float16* __restrict__ out, int n) {
  int i = blockIdx.x * 256 + threadIdx.x;
  if (i < n) out[i] = (_Float16)in[i];
}

// x [B,T,I] f32  ->  [T,B,I] f16  (row = t*B + b)
__global__ __launch_bounds__(256) void cvt_x_tbi_k(const float* __restrict__ in,
                                                   _Float16* __restrict__ out,
                                                   int Bn, int Tn, int In) {
  int idx = blockIdx.x * 256 + threadIdx.x;
  if (idx >= Bn * Tn * In) return;
  int i = idx % In;
  int r = idx / In;        // output row = t*B + b
  int b = r % Bn;
  int t = r / Bn;
  out[idx] = (_Float16)in[((size_t)b * Tn + t) * In + i];
}

__global__ __launch_bounds__(256) void bias_add_k(const float* __restrict__ a,
                                                  const float* __restrict__ b,
                                                  float* __restrict__ out, int n) {
  int i = blockIdx.x * 256 + threadIdx.x;
  if (i < n) out[i] = a[i] + b[i];
}

// ---------------------------------------------------------------- GEMM:
// C[M,N](f16) = A[M,K](f16) @ W[N,K](f16)^T + bias[N]
// Workgroup: 256 thr = 8 waves as 4(M)x2(N); wave tile 32x32 (2x2 WMMA frags).
__global__ __launch_bounds__(256) void gemm_xproj_k(
    const _Float16* __restrict__ A, const _Float16* __restrict__ W,
    const float* __restrict__ bias, _Float16* __restrict__ C,
    int M, int N, int K) {
  const int lane = threadIdx.x & 31;
  const int wave = threadIdx.x >> 5;
  const int lcol = lane & 15;
  const int hi   = lane >> 4;
  const int mw = wave & 3, nw = wave >> 2;
  const int m_base = blockIdx.y * 128 + mw * 32;
  const int n_base = blockIdx.x * 64  + nw * 32;

  AccF acc[2][2];
#pragma unroll
  for (int i = 0; i < 2; i++)
#pragma unroll
    for (int j = 0; j < 2; j++)
#pragma unroll
      for (int r = 0; r < 8; r++) acc[i][j].f[r] = 0.f;

  for (int kb = 0; kb < K; kb += 32) {
    FragH a[2], b[2];
#pragma unroll
    for (int i = 0; i < 2; i++) {
      const _Float16* p = A + (size_t)(m_base + i * 16 + lcol) * K + kb + hi * 8;
      a[i].f4[0] = *(const float4*)p;
      a[i].f4[1] = *(const float4*)(p + 16);
      const _Float16* q = W + (size_t)(n_base + i * 16 + lcol) * K + kb + hi * 8;
      b[i].f4[0] = *(const float4*)q;
      b[i].f4[1] = *(const float4*)(q + 16);
    }
#pragma unroll
    for (int i = 0; i < 2; i++)
#pragma unroll
      for (int j = 0; j < 2; j++)
        acc[i][j].v = WMMA_F16(a[i].v, b[j].v, acc[i][j].v);
  }

#pragma unroll
  for (int i = 0; i < 2; i++)
#pragma unroll
    for (int j = 0; j < 2; j++) {
      int n = n_base + j * 16 + lcol;
      float bv = bias[n];
#pragma unroll
      for (int r = 0; r < 8; r++) {
        int m = m_base + i * 16 + r + hi * 8;
        C[(size_t)m * N + n] = (_Float16)(acc[i][j].f[r] + bv);
      }
    }
}

// ---------------------------------------------------------------- LSTM recurrence
// One workgroup owns a 16-row batch tile for all T steps. h kept in LDS (f16),
// c in registers. Wave w owns gate columns {g*256 + w*16 + 0..15}, so the cell
// update is register-local. xproj slab for step t+1 is prefetched into LDS via
// async global->LDS copies overlapped with step t's WMMAs.
__global__ __launch_bounds__(512) void lstm_rec_k(
    const _Float16* __restrict__ xproj,  // [T*B, 1024], row = t*B + b
    const _Float16* __restrict__ Whh,    // [1024, 256]  (N,K) row-major
    _Float16* __restrict__ h_all,        // [T*B, 256] if store_all
    float* __restrict__ h_last,          // [B, 256]   if !store_all
    int Bn, int T, int store_all) {
  __shared__ _Float16 lds_h[16 * 256];     //  8 KB
  __shared__ _Float16 lds_xp[16 * 1024];   // 32 KB staging for one step's gates

  const int tid  = threadIdx.x;
  const int lane = tid & 31;
  const int wave = tid >> 5;   // 0..15
  const int lcol = lane & 15;
  const int hi   = lane >> 4;
  const int b0   = blockIdx.x * 16;
  const int hcol = wave * 16 + lcol;   // hidden-unit column 0..255

  float creg[8];
#pragma unroll
  for (int r = 0; r < 8; r++) creg[r] = 0.f;

  // prologue: stage slab for t=0 (rows 0*B + b0 .. +15, contiguous 32 KB)
  {
    const _Float16* src = xproj + (size_t)b0 * 1024;
#pragma unroll
    for (int p = 0; p < 4; p++) {
      int off = tid * 8 + p * 4096;  // halves; 512 thr * 16B * 4 = 32 KB
      async_copy16(src + off, &lds_xp[off]);
    }
    async_wait0();
  }
  __syncthreads();

  for (int t = 0; t < T; t++) {
    // accumulator init from staged input projection (+folded biases)
    AccF acc[4];
#pragma unroll
    for (int g = 0; g < 4; g++) {
      int n = g * 256 + hcol;
#pragma unroll
      for (int r = 0; r < 8; r++)
        acc[g].f[r] = (float)lds_xp[(r + hi * 8) * 1024 + n];
    }
    __syncthreads();  // everyone done reading lds_xp for step t

    // prefetch step t+1's slab; completes during WMMA + gate math below
    if (t + 1 < T) {
      const _Float16* src = xproj + ((size_t)(t + 1) * Bn + b0) * 1024;
#pragma unroll
      for (int p = 0; p < 4; p++) {
        int off = tid * 8 + p * 4096;
        async_copy16(src + off, &lds_xp[off]);
      }
    }

    if (t > 0) {
#pragma unroll
      for (int kb = 0; kb < 256; kb += 32) {
        FragH a;
        const _Float16* p = lds_h + lcol * 256 + kb + hi * 8;
        a.f4[0] = *(const float4*)p;
        a.f4[1] = *(const float4*)(p + 16);
#pragma unroll
        for (int g = 0; g < 4; g++) {
          FragH b;
          const _Float16* q = Whh + (size_t)(g * 256 + hcol) * 256 + kb + hi * 8;
          b.f4[0] = *(const float4*)q;
          b.f4[1] = *(const float4*)(q + 16);
          acc[g].v = WMMA_F16(a.v, b.v, acc[g].v);
        }
      }
    }

    // gate nonlinearities + cell update, all in registers
    float hv[8];
#pragma unroll
    for (int r = 0; r < 8; r++) {
      float ig = 1.f / (1.f + __expf(-acc[0].f[r]));
      float fg = 1.f / (1.f + __expf(-acc[1].f[r]));
      float gg = tanhf(acc[2].f[r]);
      float og = 1.f / (1.f + __expf(-acc[3].f[r]));
      float c  = fg * creg[r] + ig * gg;
      creg[r]  = c;
      hv[r]    = og * tanhf(c);
    }
    __syncthreads();  // all waves done reading lds_h for this step

#pragma unroll
    for (int r = 0; r < 8; r++) {
      int m = r + hi * 8;
      _Float16 hh = (_Float16)hv[r];
      lds_h[m * 256 + hcol] = hh;
      if (store_all)       h_all[((size_t)t * Bn + b0 + m) * 256 + hcol] = hh;
      else if (t == T - 1) h_last[(size_t)(b0 + m) * 256 + hcol] = hv[r];
    }
    async_wait0();    // own prefetch for t+1 has landed (overlapped with compute)
    __syncthreads();  // lds_h + lds_xp ready for next step
  }
}

// ---------------------------------------------------------------- LN + GELU + head
__global__ __launch_bounds__(256) void head_k(
    const float* __restrict__ h_last, const float* __restrict__ gamma,
    const float* __restrict__ beta, const float* __restrict__ hw,
    const float* __restrict__ hb, float* __restrict__ out) {
  __shared__ float red[256];
  const int b = blockIdx.x, t = threadIdx.x;
  float v = h_last[b * 256 + t];

  red[t] = v; __syncthreads();
  for (int s = 128; s > 0; s >>= 1) { if (t < s) red[t] += red[t + s]; __syncthreads(); }
  float mu = red[0] * (1.f / 256.f);
  __syncthreads();

  float d = v - mu;
  red[t] = d * d; __syncthreads();
  for (int s = 128; s > 0; s >>= 1) { if (t < s) red[t] += red[t + s]; __syncthreads(); }
  float var = red[0] * (1.f / 256.f);
  __syncthreads();

  float ln = d * rsqrtf(var + 1e-5f) * gamma[t] + beta[t];
  float ge = 0.5f * ln * (1.f + erff(ln * 0.70710678118654752f));  // exact GELU

  red[t] = ge * hw[t]; __syncthreads();
  for (int s = 128; s > 0; s >>= 1) { if (t < s) red[t] += red[t + s]; __syncthreads(); }
  if (t == 0) out[b] = red[0] + hb[0];
}

// ---------------------------------------------------------------- launcher
extern "C" void kernel_launch(void* const* d_in, const int* in_sizes, int n_in,
                              void* d_out, int out_size, void* d_ws, size_t ws_size,
                              hipStream_t stream) {
  const int B = 512, T = 256, I = 64, H = 256, G = 1024;  // G = 4H
  const float* x      = (const float*)d_in[0];
  const float* w_ih0  = (const float*)d_in[1];
  const float* w_hh0  = (const float*)d_in[2];
  const float* b_ih0  = (const float*)d_in[3];
  const float* b_hh0  = (const float*)d_in[4];
  const float* w_ih1  = (const float*)d_in[5];
  const float* w_hh1  = (const float*)d_in[6];
  const float* b_ih1  = (const float*)d_in[7];
  const float* b_hh1  = (const float*)d_in[8];
  const float* gamma  = (const float*)d_in[9];
  const float* beta   = (const float*)d_in[10];
  const float* head_w = (const float*)d_in[11];
  const float* head_b = (const float*)d_in[12];
  float* out = (float*)d_out;
  (void)in_sizes; (void)n_in; (void)out_size; (void)ws_size;

  char* ws = (char*)d_ws;
  size_t off = 0;
  auto alloc = [&](size_t bytes) -> void* {
    void* p = ws + off;
    off += (bytes + 255) & ~(size_t)255;
    return p;
  };
  _Float16* Wih0 = (_Float16*)alloc((size_t)G * I * 2);
  _Float16* Whh0 = (_Float16*)alloc((size_t)G * H * 2);
  _Float16* Wih1 = (_Float16*)alloc((size_t)G * H * 2);
  _Float16* Whh1 = (_Float16*)alloc((size_t)G * H * 2);
  float*    bias0 = (float*)alloc((size_t)G * 4);
  float*    bias1 = (float*)alloc((size_t)G * 4);
  _Float16* Xh    = (_Float16*)alloc((size_t)B * T * I * 2);   // [T,B,I]
  _Float16* XP    = (_Float16*)alloc((size_t)B * T * G * 2);   // [T,B,4H], reused
  _Float16* H1    = (_Float16*)alloc((size_t)B * T * H * 2);   // [T,B,H]
  float*    H2l   = (float*)alloc((size_t)B * H * 4);          // [B,H]

  auto cvt = [&](const float* in, _Float16* o, int n) {
    cvt_f32_to_f16_k<<<dim3((n + 255) / 256), dim3(256), 0, stream>>>(in, o, n);
  };
  cvt(w_ih0, Wih0, G * I);
  cvt(w_hh0, Whh0, G * H);
  cvt(w_ih1, Wih1, G * H);
  cvt(w_hh1, Whh1, G * H);
  cvt_x_tbi_k<<<dim3((B * T * I + 255) / 256), dim3(256), 0, stream>>>(x, Xh, B, T, I);
  bias_add_k<<<dim3(4), dim3(256), 0, stream>>>(b_ih0, b_hh0, bias0, G);
  bias_add_k<<<dim3(4), dim3(256), 0, stream>>>(b_ih1, b_hh1, bias1, G);

  // Layer 0: input projection (K=64), then recurrence (writes all h1)
  gemm_xproj_k<<<dim3(G / 64, (B * T) / 128), dim3(256), 0, stream>>>(
      Xh, Wih0, bias0, XP, B * T, G, I);
  lstm_rec_k<<<dim3(B / 16), dim3(512), 0, stream>>>(XP, Whh0, H1, nullptr, B, T, 1);

  // Layer 1: input projection (K=256) into same XP buffer, then recurrence
  // (only last timestep materialized, f32)
  gemm_xproj_k<<<dim3(G / 64, (B * T) / 128), dim3(256), 0, stream>>>(
      H1, Wih1, bias1, XP, B * T, G, H);
  lstm_rec_k<<<dim3(B / 16), dim3(512), 0, stream>>>(XP, Whh1, nullptr, H2l, B, T, 0);

  // LayerNorm + exact GELU + linear head
  head_k<<<dim3(B), dim3(256), 0, stream>>>(H2l, gamma, beta, head_w, head_b, out);
}
